// DirectSR_55611236548808
// MI455X (gfx1250) — matrix operand order repaired and checked by reading, not compiled, not run on hardware
//
#include <hip/hip_runtime.h>
#include <math.h>

typedef __attribute__((ext_vector_type(16))) _Float16 v16h;
typedef __attribute__((ext_vector_type(8)))  float    v8f;

#define NNODES 16384
#define NEDGES 131072
#define HEADS  4
#define HC1    512
#define HC2    256
#define CC1    128
#define CC2    64

// ---------- ordered-uint encoding for float atomicMax ----------
static __device__ __forceinline__ unsigned f_enc(float f) {
  unsigned u = __float_as_uint(f);
  return (u & 0x80000000u) ? ~u : (u | 0x80000000u);
}
static __device__ __forceinline__ float f_dec(unsigned u) {
  return __uint_as_float((u & 0x80000000u) ? (u & 0x7fffffffu) : ~u);
}
#define ENC_NEG_INF 0x007FFFFFu   // f_enc(-inf)

// ---------- init kernels ----------
__global__ void fill_f32(float* p, size_t n, float v) {
  size_t i = (size_t)blockIdx.x * blockDim.x + threadIdx.x;
  if (i < n) p[i] = v;
}
__global__ void fill_u32(unsigned* p, size_t n, unsigned v) {
  size_t i = (size_t)blockIdx.x * blockDim.x + threadIdx.x;
  if (i < n) p[i] = v;
}

// =====================================================================
// Operand packing into WMMA-fragment order (f32 -> f16, done once per use)
//
// A fragment (16x32 f16), lane l: m = l&15, hi = l>>4,
//   element i<8 : K = hi*8 + i        (contiguous run)
//   element i>=8: K = hi*8 + 16 + i-8 (contiguous run)
// B fragment (32x16 f16), lane l: n = l&15, element i: same K as A element i.
// Packed layout: [tile][lane][16 halfs]  (32B per lane, contiguous)
// =====================================================================

// X[M, Ci] row-major -> A-fragment packed; tile = rt*KT + kt, KT = Ci/32
__global__ __launch_bounds__(256) void pack_a(const float* __restrict__ X,
                                              _Float16* __restrict__ Ap, int Ci)
{
  int idx  = blockIdx.x * 256 + threadIdx.x;
  int lane = idx & 31, tile = idx >> 5;
  int KT = Ci >> 5;
  int rt = tile / KT, kt = tile % KT;
  int m = lane & 15, hi = lane >> 4;
  const float* p = X + (size_t)(rt * 16 + m) * Ci + kt * 32 + hi * 8;
  v16h v;
#pragma unroll
  for (int i = 0; i < 8; ++i) v[i] = (_Float16)p[i];
#pragma unroll
  for (int i = 0; i < 8; ++i) v[8 + i] = (_Float16)p[16 + i];
  *(v16h*)(Ap + (size_t)tile * 512 + lane * 16) = v;
}

// S[K_total, M] row-major, pack transpose(S) (A[m,K] = S[K,m]) into A-fragment
// order; tile = rt*KT + kt, KT = K_total/32. (Used for Gram: A = h^T)
__global__ __launch_bounds__(256) void pack_at(const float* __restrict__ S,
                                               _Float16* __restrict__ Ap,
                                               int Ktot, int M)
{
  int idx  = blockIdx.x * 256 + threadIdx.x;
  int lane = idx & 31, tile = idx >> 5;
  int KT = Ktot >> 5;
  int rt = tile / KT, kt = tile % KT;
  int m = lane & 15, hi = lane >> 4;
  const float* p = S + (size_t)(kt * 32 + hi * 8) * M + rt * 16 + m;
  v16h v;
#pragma unroll
  for (int i = 0; i < 8; ++i) v[i] = (_Float16)p[(size_t)i * M];
#pragma unroll
  for (int i = 0; i < 8; ++i) v[8 + i] = (_Float16)p[(size_t)(16 + i) * M];
  *(v16h*)(Ap + (size_t)tile * 512 + lane * 16) = v;
}

// W[Ci, Co] row-major -> B-fragment packed; tile = kt*CT + ct, CT = Co/16
__global__ __launch_bounds__(256) void pack_b(const float* __restrict__ W,
                                              _Float16* __restrict__ Bp, int Co)
{
  int idx  = blockIdx.x * 256 + threadIdx.x;
  int lane = idx & 31, tile = idx >> 5;
  int CT = Co >> 4;
  int kt = tile / CT, ct = tile % CT;
  int n = lane & 15, hi = lane >> 4;
  const float* p = W + (size_t)(kt * 32 + hi * 8) * Co + ct * 16 + n;
  v16h v;
#pragma unroll
  for (int i = 0; i < 8; ++i) v[i] = (_Float16)p[(size_t)i * Co];
#pragma unroll
  for (int i = 0; i < 8; ++i) v[8 + i] = (_Float16)p[(size_t)(16 + i) * Co];
  *(v16h*)(Bp + (size_t)tile * 512 + lane * 16) = v;
}

// =====================================================================
// Packed WMMA GEMM: Y[M,Co] = unpack(Ap) @ unpack(Bp) + bias
// 8 waves/block; each wave: 2 row tiles x 4 col tiles (8 accumulators).
// Ping-pong prefetch: k-step kt+1 fragments are loaded before the WMMAs of
// kt issue, so loads overlap the matrix pipe (partial loadcnt waits).
// NOTE: prefetches one k-step past the end -> packed B needs 1 k-row pad.
// =====================================================================
#define WMMA_F16(A_, B_, C_) \
  __builtin_amdgcn_wmma_f32_16x16x32_f16(false, (A_), false, (B_), (short)0, (C_), false, false)

__global__ __launch_bounds__(256) void gemm_packed(
    const _Float16* __restrict__ Ap, const _Float16* __restrict__ Bp,
    const float* __restrict__ bias, float* __restrict__ Y,
    int KT, int CT, int Co)
{
  const int wave = threadIdx.x >> 5;
  const int lane = threadIdx.x & 31;
  const int groups = CT >> 2;               // col-tile groups of 4
  const int wtile = blockIdx.x * 8 + wave;
  const int rt  = (wtile / groups) * 2;     // row-tile pair
  const int ct0 = (wtile % groups) << 2;

  const v16h* ap0 = (const v16h*)Ap + (size_t)rt * KT * 32 + lane;
  const v16h* ap1 = ap0 + (size_t)KT * 32;
  const v16h* bp  = (const v16h*)Bp + (size_t)ct0 * 32 + lane;
  const size_t bstep = (size_t)CT * 32;

  v8f acc0 = {}, acc1 = {}, acc2 = {}, acc3 = {};
  v8f acc4 = {}, acc5 = {}, acc6 = {}, acc7 = {};

  // preload k-step 0
  v16h ax0 = ap0[0], ax1 = ap1[0];
  v16h bx0 = bp[0], bx1 = bp[32], bx2 = bp[64], bx3 = bp[96];
  ap0 += 32; ap1 += 32; bp += bstep;

  for (int kt = 0; kt < KT; kt += 2) {
    // prefetch odd k-step
    v16h ay0 = ap0[0], ay1 = ap1[0];
    v16h by0 = bp[0], by1 = bp[32], by2 = bp[64], by3 = bp[96];
    ap0 += 32; ap1 += 32; bp += bstep;

    acc0 = WMMA_F16(ax0, bx0, acc0);
    acc1 = WMMA_F16(ax0, bx1, acc1);
    acc2 = WMMA_F16(ax0, bx2, acc2);
    acc3 = WMMA_F16(ax0, bx3, acc3);
    acc4 = WMMA_F16(ax1, bx0, acc4);
    acc5 = WMMA_F16(ax1, bx1, acc5);
    acc6 = WMMA_F16(ax1, bx2, acc6);
    acc7 = WMMA_F16(ax1, bx3, acc7);

    // prefetch next even k-step (last iteration reads into the pad)
    ax0 = ap0[0]; ax1 = ap1[0];
    bx0 = bp[0]; bx1 = bp[32]; bx2 = bp[64]; bx3 = bp[96];
    ap0 += 32; ap1 += 32; bp += bstep;

    acc0 = WMMA_F16(ay0, by0, acc0);
    acc1 = WMMA_F16(ay0, by1, acc1);
    acc2 = WMMA_F16(ay0, by2, acc2);
    acc3 = WMMA_F16(ay0, by3, acc3);
    acc4 = WMMA_F16(ay1, by0, acc4);
    acc5 = WMMA_F16(ay1, by1, acc5);
    acc6 = WMMA_F16(ay1, by2, acc6);
    acc7 = WMMA_F16(ay1, by3, acc7);
  }

  const int hi = lane >> 4;
  const int n  = lane & 15;
  v8f accs[8] = {acc0, acc1, acc2, acc3, acc4, acc5, acc6, acc7};
#pragma unroll
  for (int rr = 0; rr < 2; ++rr) {
#pragma unroll
    for (int t = 0; t < 4; ++t) {
      int col = (ct0 + t) * 16 + n;
      float bs = bias ? bias[col] : 0.0f;
#pragma unroll
      for (int r = 0; r < 8; ++r) {
        int row = (rt + rr) * 16 + r + hi * 8;   // C/D f32 16x16 layout
        Y[(size_t)row * Co + col] = accs[rr * 4 + t][r] + bs;
      }
    }
  }
}

// ---------- per-edge attention logits + segment max ----------
__global__ void attn_logits(const float* __restrict__ q, const float* __restrict__ k,
                            const float* __restrict__ We, const float* __restrict__ ea,
                            const int* __restrict__ src, const int* __restrict__ dst,
                            float* __restrict__ alpha, unsigned* __restrict__ amax,
                            int C, float scale)
{
  int t = blockIdx.x * blockDim.x + threadIdx.x;
  if (t >= NEDGES * HEADS) return;
  int e = t >> 2;
  int h = t & 3;
  int s = src[e], d = dst[e];
  float eav = ea[e];
  const float* qp = q + (size_t)d * (HEADS * C) + h * C;
  const float* kp = k + (size_t)s * (HEADS * C) + h * C;
  const float* wp = We + h * C;
  float acc = 0.f;
  for (int c = 0; c < C; ++c)
    acc += qp[c] * (kp[c] + eav * wp[c]);
  acc *= scale;
  alpha[t] = acc;
  atomicMax(&amax[d * HEADS + h], f_enc(acc));
}

// ---------- segment softmax denominator ----------
__global__ void attn_denom(const int* __restrict__ dst, const float* __restrict__ alpha,
                           const unsigned* __restrict__ amax, float* __restrict__ denom)
{
  int t = blockIdx.x * blockDim.x + threadIdx.x;
  if (t >= NEDGES * HEADS) return;
  int e = t >> 2;
  int h = t & 3;
  int d = dst[e];
  float ex = __expf(alpha[t] - f_dec(amax[d * HEADS + h]));
  atomicAdd(&denom[d * HEADS + h], ex);
}

// ---------- weighted scatter: out[dst] += a * (v[src] + edge_emb) ----------
__global__ void attn_scatter(const float* __restrict__ v, const float* __restrict__ We,
                             const float* __restrict__ ea, const int* __restrict__ src,
                             const int* __restrict__ dst, const float* __restrict__ alpha,
                             const unsigned* __restrict__ amax, const float* __restrict__ denom,
                             float* __restrict__ out, int C)
{
  __shared__ float a_sh[HEADS];
  int e = blockIdx.x;
  int ch = threadIdx.x;
  int d = dst[e];
  if (ch < HEADS) {
    float ex = __expf(alpha[e * HEADS + ch] - f_dec(amax[d * HEADS + ch]));
    a_sh[ch] = ex / (denom[d * HEADS + ch] + 1e-16f);
  }
  __syncthreads();
  int s = src[e];
  int h = ch / C;
  float val = a_sh[h] * (v[(size_t)s * (HEADS * C) + ch] + ea[e] * We[ch]);
  atomicAdd(&out[(size_t)d * (HEADS * C) + ch], val);
}

// ---------- beta gate + skip combine ----------
__global__ __launch_bounds__(128) void beta_combine(
    const float* __restrict__ out, const float* __restrict__ xr,
    const float* __restrict__ Wb, float* __restrict__ hdst, int HC)
{
  __shared__ float red[128];
  int n = blockIdx.x;
  int t = threadIdx.x;
  const float* op = out + (size_t)n * HC;
  const float* xp = xr + (size_t)n * HC;
  float s = 0.f;
  for (int c = t; c < HC; c += 128) {
    float o = op[c], x = xp[c];
    s += o * Wb[c] + x * Wb[HC + c] + (o - x) * Wb[2 * HC + c];
  }
  red[t] = s;
  __syncthreads();
  for (int w = 64; w > 0; w >>= 1) {
    if (t < w) red[t] += red[t + w];
    __syncthreads();
  }
  float beta = 1.f / (1.f + __expf(-red[0]));
  for (int c = t; c < HC; c += 128) {
    float o = op[c], x = xp[c];
    hdst[(size_t)n * HC + c] = beta * x + (1.f - beta) * o;
  }
}

// ---------- GraphNorm reductions (one block per channel) ----------
__global__ __launch_bounds__(256) void col_mean(const float* __restrict__ x,
                                                float* __restrict__ mean, int HC)
{
  __shared__ float red[256];
  int ch = blockIdx.x, t = threadIdx.x;
  float s = 0.f;
  for (int i = t; i < NNODES; i += 256) s += x[(size_t)i * HC + ch];
  red[t] = s;
  __syncthreads();
  for (int w = 128; w > 0; w >>= 1) {
    if (t < w) red[t] += red[t + w];
    __syncthreads();
  }
  if (t == 0) mean[ch] = red[0] * (1.f / NNODES);
}

__global__ __launch_bounds__(256) void col_var(const float* __restrict__ x,
                                               const float* __restrict__ mean,
                                               const float* __restrict__ ms,
                                               float* __restrict__ var, int HC)
{
  __shared__ float red[256];
  int ch = blockIdx.x, t = threadIdx.x;
  float mu = mean[ch] * ms[ch];
  float s = 0.f;
  for (int i = t; i < NNODES; i += 256) {
    float d = x[(size_t)i * HC + ch] - mu;
    s += d * d;
  }
  red[t] = s;
  __syncthreads();
  for (int w = 128; w > 0; w >>= 1) {
    if (t < w) red[t] += red[t + w];
    __syncthreads();
  }
  if (t == 0) var[ch] = red[0] * (1.f / NNODES);
}

__global__ void norm_relu(float* __restrict__ x, const float* __restrict__ mean,
                          const float* __restrict__ ms, const float* __restrict__ var,
                          const float* __restrict__ w, const float* __restrict__ b, int HC)
{
  size_t i = (size_t)blockIdx.x * blockDim.x + threadIdx.x;
  if (i >= (size_t)NNODES * HC) return;
  int ch = (int)(i % HC);
  float d = x[i] - mean[ch] * ms[ch];
  float y = d * rsqrtf(var[ch] + 1e-5f) * w[ch] + b[ch];
  x[i] = fmaxf(y, 0.f);
}

// ---------- single-block min/max of G ----------
__global__ __launch_bounds__(256) void minmax256(const float* __restrict__ G,
                                                 float* __restrict__ mm)
{
  __shared__ float rmin[256], rmax[256];
  int t = threadIdx.x;
  float mn = 3.4e38f, mx = -3.4e38f;
  for (int i = t; i < 65536; i += 256) {
    float v = G[i];
    mn = fminf(mn, v);
    mx = fmaxf(mx, v);
  }
  rmin[t] = mn; rmax[t] = mx;
  __syncthreads();
  for (int w = 128; w > 0; w >>= 1) {
    if (t < w) { rmin[t] = fminf(rmin[t], rmin[t + w]); rmax[t] = fmaxf(rmax[t], rmax[t + w]); }
    __syncthreads();
  }
  if (t == 0) { mm[0] = rmin[0]; mm[1] = rmax[0]; }
}

__global__ void finalize_out(const float* __restrict__ G, const float* __restrict__ mm,
                             const float* __restrict__ tgt, float* __restrict__ out)
{
  int i = blockIdx.x * blockDim.x + threadIdx.x;  // 65536 threads
  float mn = mm[0], mx = mm[1];
  out[i] = (G[i] - mn) / (mx - mn + 1e-8f);
  out[65536 + i] = tgt[i];
}

// =====================================================================
extern "C" void kernel_launch(void* const* d_in, const int* in_sizes, int n_in,
                              void* d_out, int out_size, void* d_ws, size_t ws_size,
                              hipStream_t stream) {
  (void)in_sizes; (void)n_in; (void)out_size; (void)ws_size;
  const float* x   = (const float*)d_in[0];
  const int*   ei  = (const int*)d_in[1];
  const float* ea  = (const float*)d_in[2];
  const float* tgt = (const float*)d_in[3];
  // conv1 params
  const float *Wq1 = (const float*)d_in[4],  *bq1 = (const float*)d_in[5];
  const float *Wk1 = (const float*)d_in[6],  *bk1 = (const float*)d_in[7];
  const float *Wv1 = (const float*)d_in[8],  *bv1 = (const float*)d_in[9];
  const float *We1 = (const float*)d_in[10];
  const float *Ws1 = (const float*)d_in[11], *bs1 = (const float*)d_in[12];
  const float *Wb1 = (const float*)d_in[13];
  const float *g1w = (const float*)d_in[14], *g1b = (const float*)d_in[15], *g1m = (const float*)d_in[16];
  // conv2 params
  const float *Wq2 = (const float*)d_in[17], *bq2 = (const float*)d_in[18];
  const float *Wk2 = (const float*)d_in[19], *bk2 = (const float*)d_in[20];
  const float *Wv2 = (const float*)d_in[21], *bv2 = (const float*)d_in[22];
  const float *We2 = (const float*)d_in[23];
  const float *Ws2 = (const float*)d_in[24], *bs2 = (const float*)d_in[25];
  const float *Wb2 = (const float*)d_in[26];
  const float *g2w = (const float*)d_in[27], *g2b = (const float*)d_in[28], *g2m = (const float*)d_in[29];

  // workspace layout (float units)
  float* ws = (float*)d_ws;
  const size_t NM = NNODES;
  float* q   = ws;
  float* kk  = q  + NM * HC1;
  float* v   = kk + NM * HC1;
  float* xr  = v  + NM * HC1;
  float* out = xr + NM * HC1;   // attention accumulator
  float* h   = out + NM * HC1;  // h1, then h2
  float* alpha = h + NM * HC1;                                  // [E,H]
  unsigned* amax = (unsigned*)(alpha + (size_t)NEDGES * HEADS); // [N,H]
  float* denom = (float*)(amax + NM * HEADS);                   // [N,H]
  float* mean  = denom + NM * HEADS;                            // [512]
  float* var   = mean + 512;                                    // [512]
  float* G     = var + 512;                                     // [256,256]
  float* mm    = G + 65536;                                     // min,max
  _Float16* Ap = (_Float16*)(mm + 16);         // packed A: 8M halfs (16MB)
  _Float16* Bp = Ap + (size_t)NM * HC1;        // packed B: 4x 512*256 halfs (+ pad)
  // Gram packs reuse the Ap region (free after conv2 GEMMs)
  _Float16* ApG = Ap;                          // 256 x 16384 halfs
  _Float16* BpG = Ap + (size_t)256 * NNODES;   // 16384 x 256 halfs

  const int* src = ei;
  const int* dst = ei + NEDGES;

  const dim3 b256(256);
  const int eh_blocks = (NEDGES * HEADS) / 256;

  // =========================== conv1 ===========================
  {
    const int KT = CC1 / 32, CT = HC1 / 16;                 // 4, 32
    const size_t wsz = (size_t)CC1 * HC1;                   // 65536
    pack_a<<<(NNODES / 16) * KT * 32 / 256, b256, 0, stream>>>(x, Ap, CC1);
    pack_b<<<KT * CT * 32 / 256, b256, 0, stream>>>(Wq1, Bp,           HC1);
    pack_b<<<KT * CT * 32 / 256, b256, 0, stream>>>(Wk1, Bp + wsz,     HC1);
    pack_b<<<KT * CT * 32 / 256, b256, 0, stream>>>(Wv1, Bp + 2 * wsz, HC1);
    pack_b<<<KT * CT * 32 / 256, b256, 0, stream>>>(Ws1, Bp + 3 * wsz, HC1);
    const int gblocks = (NNODES / 32) * (CT / 4) / 8;       // 512
    gemm_packed<<<gblocks, b256, 0, stream>>>(Ap, Bp,           bq1, q,  KT, CT, HC1);
    gemm_packed<<<gblocks, b256, 0, stream>>>(Ap, Bp + wsz,     bk1, kk, KT, CT, HC1);
    gemm_packed<<<gblocks, b256, 0, stream>>>(Ap, Bp + 2 * wsz, bv1, v,  KT, CT, HC1);
    gemm_packed<<<gblocks, b256, 0, stream>>>(Ap, Bp + 3 * wsz, bs1, xr, KT, CT, HC1);

    fill_f32<<<(unsigned)((NM * HC1 + 255) / 256), b256, 0, stream>>>(out, NM * HC1, 0.f);
    fill_f32<<<(unsigned)((NM * HEADS + 255) / 256), b256, 0, stream>>>(denom, NM * HEADS, 0.f);
    fill_u32<<<(unsigned)((NM * HEADS + 255) / 256), b256, 0, stream>>>(amax, NM * HEADS, ENC_NEG_INF);

    attn_logits<<<eh_blocks, b256, 0, stream>>>(q, kk, We1, ea, src, dst, alpha, amax,
                                                CC1, 0.08838834764831845f); // 1/sqrt(128)
    attn_denom<<<eh_blocks, b256, 0, stream>>>(dst, alpha, amax, denom);
    attn_scatter<<<NEDGES, HC1, 0, stream>>>(v, We1, ea, src, dst, alpha, amax, denom, out, CC1);
    beta_combine<<<NNODES, 128, 0, stream>>>(out, xr, Wb1, h, HC1);

    col_mean<<<HC1, b256, 0, stream>>>(h, mean, HC1);
    col_var<<<HC1, b256, 0, stream>>>(h, mean, g1m, var, HC1);
    norm_relu<<<(unsigned)((NM * HC1 + 255) / 256), b256, 0, stream>>>(h, mean, g1m, var, g1w, g1b, HC1);
  }

  // =========================== conv2 ===========================
  {
    const int KT = HC1 / 32, CT = HC2 / 16;                 // 16, 16
    const size_t wsz = (size_t)HC1 * HC2;                   // 131072
    pack_a<<<(NNODES / 16) * KT * 32 / 256, b256, 0, stream>>>(h, Ap, HC1);
    pack_b<<<KT * CT * 32 / 256, b256, 0, stream>>>(Wq2, Bp,           HC2);
    pack_b<<<KT * CT * 32 / 256, b256, 0, stream>>>(Wk2, Bp + wsz,     HC2);
    pack_b<<<KT * CT * 32 / 256, b256, 0, stream>>>(Wv2, Bp + 2 * wsz, HC2);
    pack_b<<<KT * CT * 32 / 256, b256, 0, stream>>>(Ws2, Bp + 3 * wsz, HC2);
    const int gblocks = (NNODES / 32) * (CT / 4) / 8;       // 256
    gemm_packed<<<gblocks, b256, 0, stream>>>(Ap, Bp,           bq2, q,  KT, CT, HC2);
    gemm_packed<<<gblocks, b256, 0, stream>>>(Ap, Bp + wsz,     bk2, kk, KT, CT, HC2);
    gemm_packed<<<gblocks, b256, 0, stream>>>(Ap, Bp + 2 * wsz, bv2, v,  KT, CT, HC2);
    gemm_packed<<<gblocks, b256, 0, stream>>>(Ap, Bp + 3 * wsz, bs2, xr, KT, CT, HC2);

    fill_f32<<<(unsigned)((NM * HC2 + 255) / 256), b256, 0, stream>>>(out, NM * HC2, 0.f);
    fill_f32<<<(unsigned)((NM * HEADS + 255) / 256), b256, 0, stream>>>(denom, NM * HEADS, 0.f);
    fill_u32<<<(unsigned)((NM * HEADS + 255) / 256), b256, 0, stream>>>(amax, NM * HEADS, ENC_NEG_INF);

    attn_logits<<<eh_blocks, b256, 0, stream>>>(q, kk, We2, ea, src, dst, alpha, amax,
                                                CC2, 0.125f); // 1/sqrt(64)
    attn_denom<<<eh_blocks, b256, 0, stream>>>(dst, alpha, amax, denom);
    attn_scatter<<<NEDGES, HC2, 0, stream>>>(v, We2, ea, src, dst, alpha, amax, denom, out, CC2);
    beta_combine<<<NNODES, 128, 0, stream>>>(out, xr, Wb2, h, HC2);

    col_mean<<<HC2, b256, 0, stream>>>(h, mean, HC2);
    col_var<<<HC2, b256, 0, stream>>>(h, mean, g2m, var, HC2);
    norm_relu<<<(unsigned)((NM * HC2 + 255) / 256), b256, 0, stream>>>(h, mean, g2m, var, g2w, g2b, HC2);
  }

  // ======================= Gram + normalize =====================
  {
    const int KT = NNODES / 32;                             // 512
    const int CT = HC2 / 16;                                // 16
    pack_at<<<16 * KT * 32 / 256, b256, 0, stream>>>(h, ApG, NNODES, HC2); // A = h^T
    pack_b<<<KT * CT * 32 / 256, b256, 0, stream>>>(h, BpG, HC2);          // B = h
    const int gblocks = (256 / 32) * (CT / 4) / 8;          // 4
    gemm_packed<<<gblocks, b256, 0, stream>>>(ApG, BpG, (const float*)nullptr, G, KT, CT, HC2);
  }
  minmax256<<<1, b256, 0, stream>>>(G, mm);
  finalize_out<<<256, b256, 0, stream>>>(G, mm, tgt, (float*)d_out);
}
